// AttentionUpscaling_35545149341953
// MI455X (gfx1250) — matrix operand order repaired and checked by reading, not compiled, not run on hardware
//
#include <hip/hip_runtime.h>
#include <cstdint>

typedef __bf16 bf16_t;
typedef __attribute__((ext_vector_type(8)))  __bf16 v8bf;
typedef __attribute__((ext_vector_type(16))) __bf16 v16bf;
typedef __attribute__((ext_vector_type(8)))  float  v8f;

#define BATCH 2
#define NPATCH 1024          // N = 32*32
#define DHF 2048
#define DIN 512              // C*POOL*POOL
#define DEMB 128
#define HID 128
#define TOPK 16
#define MROWS (BATCH * NPATCH)   // 2048

// ---------------------------------------------------------------------------
// Kernel 1: per-row top-16 of hr_attn[B,N,N] via wave32 shuffle argmax.
// Outputs prior (sorted desc), log(prior), idx, and {top1, margin, mass}.
// ---------------------------------------------------------------------------
__global__ __launch_bounds__(256) void topk_kernel(
    const float* __restrict__ attn, float* __restrict__ prior,
    float* __restrict__ logp, int* __restrict__ idxb, float* __restrict__ scal)
{
    int m = blockIdx.x * (blockDim.x >> 5) + (threadIdx.x >> 5);
    int lane = threadIdx.x & 31;
    const float* row = attn + (size_t)m * 1024;
    float vals[32];
#pragma unroll
    for (int t = 0; t < 32; t++) vals[t] = row[t * 32 + lane];

    float mass = 0.f, p0 = 0.f, p1 = 0.f;
    for (int p = 0; p < TOPK; p++) {
        float bv = -1e30f; int bj = 0;
#pragma unroll
        for (int t = 0; t < 32; t++) {
            int j = t * 32 + lane;
            if (vals[t] > bv || (vals[t] == bv && j < bj)) { bv = vals[t]; bj = j; }
        }
#pragma unroll
        for (int off = 16; off >= 1; off >>= 1) {
            float ov = __shfl_xor(bv, off, 32);
            int   oj = __shfl_xor(bj, off, 32);
            if (ov > bv || (ov == bv && oj < bj)) { bv = ov; bj = oj; }
        }
        if ((bj & 31) == lane) {           // owner lane clears the winner
            int tt = bj >> 5;
#pragma unroll
            for (int t = 0; t < 32; t++) if (t == tt) vals[t] = -1e30f;
        }
        mass += bv;
        if (p == 0) p0 = bv;
        if (p == 1) p1 = bv;
        if (lane == 0) {
            prior[m * TOPK + p] = bv;
            idxb [m * TOPK + p] = bj;
            logp [m * TOPK + p] = __logf(fmaxf(bv, 1e-8f));
        }
    }
    if (lane == 0) { scal[m*3+0] = p0; scal[m*3+1] = p0 - p1; scal[m*3+2] = mass; }
}

// ---------------------------------------------------------------------------
// Kernel 2: weight prep. bf16-convert + transpose to [N][K], and build the
// W1 decomposition: Aq = W1[0:128]+W1[256:384], Ak = W1[128:256]-W1[256:384],
// Ah = W1[384:512], w1row = W1[512].
// ---------------------------------------------------------------------------
__global__ __launch_bounds__(256) void prep_kernel(
    const float* __restrict__ Wq, const float* __restrict__ Wk,
    const float* __restrict__ Wg, const float* __restrict__ W1,
    bf16_t* __restrict__ WqT, bf16_t* __restrict__ WkT, bf16_t* __restrict__ WgT,
    bf16_t* __restrict__ AqT, bf16_t* __restrict__ AkT, bf16_t* __restrict__ AhT,
    float* __restrict__ w1row)
{
    int i = blockIdx.x * blockDim.x + threadIdx.x;
    if (i < 65536)  { int k = i & 511, n = i >> 9; WkT[n*512+k] = (bf16_t)Wk[k*128+n]; return; }
    i -= 65536;
    if (i < 65536)  { int k = i & 511, n = i >> 9; WqT[n*512+k] = (bf16_t)Wq[k*128+n]; return; }
    i -= 65536;
    if (i < 262144) { int k = i & 127, n = i >> 7; WgT[n*128+k] = (bf16_t)Wg[k*2048+n]; return; }
    i -= 262144;
    if (i < 16384)  { int k = i & 127, n = i >> 7;
                      AqT[n*128+k] = (bf16_t)(W1[k*128+n] + W1[(256+k)*128+n]); return; }
    i -= 16384;
    if (i < 16384)  { int k = i & 127, n = i >> 7;
                      AkT[n*128+k] = (bf16_t)(W1[(128+k)*128+n] - W1[(256+k)*128+n]); return; }
    i -= 16384;
    if (i < 16384)  { int k = i & 127, n = i >> 7;
                      AhT[n*128+k] = (bf16_t)W1[(384+k)*128+n]; return; }
    i -= 16384;
    if (i < 128)    w1row[i] = W1[512*128 + i];
}

// ---------------------------------------------------------------------------
// Kernel 3: pooled descriptors (8x8 -> 4x4 block mean) for hr & base,
// emitted as bf16 GEMM-A matrices [M=2048][K=512]. Coalesced over n.
// ---------------------------------------------------------------------------
__global__ __launch_bounds__(256) void desc_kernel(
    const float* __restrict__ hr, const float* __restrict__ base,
    bf16_t* __restrict__ dhr, bf16_t* __restrict__ dbase)
{
    int bx = blockIdx.x;                         // b*512 + din
    int n  = blockIdx.y * blockDim.x + threadIdx.x;
    int b  = bx >> 9, din = bx & 511;
    int c = din >> 4, pY = (din >> 2) & 3, pX = din & 3;
    int dhf = c * 64 + (2 * pY) * 8 + 2 * pX;
    size_t o = ((size_t)(b * DHF + dhf)) * 1024 + n;
    size_t dsto = ((size_t)(b * NPATCH + n)) * DIN + din;
    {
        const float* p = hr + o;
        float s = p[0] + p[1024] + p[8 * 1024] + p[9 * 1024];
        dhr[dsto] = (bf16_t)(s * 0.25f);
    }
    {
        const float* p = base + o;
        float s = p[0] + p[1024] + p[8 * 1024] + p[9 * 1024];
        dbase[dsto] = (bf16_t)(s * 0.25f);
    }
}

// ---------------------------------------------------------------------------
// bf16 WMMA GEMM (streaming): C[M][N] = A[M][K] * B (given Bt[N][K]) + bias.
// One wave per 16x16 C tile, K stepped by 32 through v_wmma_f32_16x16x32_bf16.
// Used for the small-N GEMMs (embed / qW / kW) where A is read only once.
// ---------------------------------------------------------------------------
__global__ __launch_bounds__(256) void wmma_gemm_plain(
    const bf16_t* __restrict__ A, const bf16_t* __restrict__ Bt,
    int M, int N, int K,
    const float* __restrict__ bias,
    float* __restrict__ outF, bf16_t* __restrict__ outB)
{
    int wid  = blockIdx.x * (blockDim.x >> 5) + (threadIdx.x >> 5);
    int lane = threadIdx.x & 31;
    int ntiles = N >> 4;
    int tm = wid / ntiles, tn = wid % ntiles;
    if (tm * 16 >= M) return;                    // wave-uniform
    int m0 = tm * 16, n0 = tn * 16;
    int la = lane & 15, hi = lane >> 4;

    const bf16_t* arow = A  + (size_t)(m0 + la) * K + hi * 16;
    const bf16_t* brow = Bt + (size_t)(n0 + la) * K + hi * 16;

    v8f acc = {};
    for (int k0 = 0; k0 < K; k0 += 32) {
        v8bf a_lo = *(const v8bf*)(arow + k0);
        v8bf a_hi = *(const v8bf*)(arow + k0 + 8);
        v16bf av = __builtin_shufflevector(a_lo, a_hi,
                        0,1,2,3,4,5,6,7,8,9,10,11,12,13,14,15);
        v16bf bv = *(const v16bf*)(brow + k0);
        acc = __builtin_amdgcn_wmma_f32_16x16x32_bf16(
                  false, av, false, bv, (short)0, acc, false, false);
    }

    int ncol = n0 + la;
    float bn = bias ? bias[ncol] : 0.0f;
#pragma unroll
    for (int v = 0; v < 8; v++) {
        int m = m0 + v + 8 * hi;
        float val = acc[v] + bn;
        if (outF) outF[(size_t)m * N + ncol] = val;
        if (outB) outB[(size_t)m * N + ncol] = (bf16_t)val;
    }
}

// ---------------------------------------------------------------------------
// Gate GEMM, LDS-staged via CDNA5 async-to-LDS (ASYNCcnt path).
// C[2048][2048] = q_bf[2048][128] @ Wg' (Bt = WgT[2048][128]), then
// sigmoid(acc + bg + rank-3 scalar features). Block = 128x128 macro-tile:
// A tile (32KB) and B tile (32KB) are contiguous (K=128 spans the row) and
// are staged once with global_load_async_to_lds_b128, then all fragments are
// served from LDS (ds_load_b128) instead of 128x-redundant global reads.
// ---------------------------------------------------------------------------
__global__ __launch_bounds__(256) void gate_gemm_lds(
    const bf16_t* __restrict__ A, const bf16_t* __restrict__ Bt,
    const float* __restrict__ bias, float* __restrict__ outF,
    const float* __restrict__ scal, const float* __restrict__ wgTail)
{
    __shared__ __align__(64) bf16_t As[128 * 128];   // 32 KB
    __shared__ __align__(64) bf16_t Bs[128 * 128];   // 32 KB

    int tid = threadIdx.x;
    int m0 = blockIdx.x * 128;
    int n0 = blockIdx.y * 128;

    // ---- async stage: 64 KB total, 16 B per lane per instruction ----
    {
        const char* gA = (const char*)(A  + (size_t)m0 * 128);
        const char* gB = (const char*)(Bt + (size_t)n0 * 128);
        unsigned ldsA0 = (unsigned)(uintptr_t)(void*)As;
        unsigned ldsB0 = (unsigned)(uintptr_t)(void*)Bs;
#pragma unroll
        for (int i = 0; i < 8; i++) {
            unsigned off = (unsigned)(tid + 256 * i) * 16u;   // 0..32767
            unsigned la32 = ldsA0 + off;
            unsigned lb32 = ldsB0 + off;
            unsigned long long ga = (unsigned long long)(gA + off);
            unsigned long long gb = (unsigned long long)(gB + off);
            asm volatile("global_load_async_to_lds_b128 %0, %1, off"
                         :: "v"(la32), "v"(ga) : "memory");
            asm volatile("global_load_async_to_lds_b128 %0, %1, off"
                         :: "v"(lb32), "v"(gb) : "memory");
        }
        asm volatile("s_wait_asynccnt 0" ::: "memory");
    }
    __syncthreads();

    int w    = tid >> 5;                 // wave id: 16-row strip of the tile
    int lane = tid & 31;
    int la = lane & 15, hi = lane >> 4;

    // A fragments for this wave's 16 rows (4 K-steps), loaded once from LDS.
    v16bf afr[4];
    {
        const bf16_t* arow = As + (size_t)(w * 16 + la) * 128 + hi * 16;
#pragma unroll
        for (int ks = 0; ks < 4; ks++) {
            v8bf a_lo = *(const v8bf*)(arow + ks * 32);
            v8bf a_hi = *(const v8bf*)(arow + ks * 32 + 8);
            afr[ks] = __builtin_shufflevector(a_lo, a_hi,
                          0,1,2,3,4,5,6,7,8,9,10,11,12,13,14,15);
        }
    }

#pragma unroll
    for (int jt = 0; jt < 8; jt++) {     // 8 N-tiles of 16 columns
        v8f acc = {};
        const bf16_t* brow = Bs + (size_t)(jt * 16 + la) * 128 + hi * 16;
#pragma unroll
        for (int ks = 0; ks < 4; ks++) {
            v16bf bv = *(const v16bf*)(brow + ks * 32);
            acc = __builtin_amdgcn_wmma_f32_16x16x32_bf16(
                      false, afr[ks], false, bv, (short)0, acc, false, false);
        }
        int ncol = n0 + jt * 16 + la;
        float bn = bias[ncol];
        float w0 = wgTail[ncol], w1 = wgTail[2048 + ncol], w2 = wgTail[4096 + ncol];
#pragma unroll
        for (int v = 0; v < 8; v++) {
            int m = m0 + w * 16 + v + 8 * hi;
            const float* s = scal + m * 3;
            float val = acc[v] + bn + s[0] * w0 + s[1] * w1 + s[2] * w2;
            outF[(size_t)m * DHF + ncol] = 1.0f / (1.0f + __expf(-val));
        }
    }
}

// ---------------------------------------------------------------------------
// Kernel: rescorer. One wave per (b,n): a 16x16 output tile's rows are exactly
// the 16 top-k candidates. hidden = qWb1[n] + kW[idx] + (q .* key[idx])@Ah
//                               + prior*w1row ; res = relu(hidden)@W2.
// Then softmax(log prior + res) -> weights.
// ---------------------------------------------------------------------------
__global__ __launch_bounds__(256) void rescore_kernel(
    const bf16_t* __restrict__ qbf, const bf16_t* __restrict__ kbf,
    const bf16_t* __restrict__ AhT,
    const float* __restrict__ qWb1, const float* __restrict__ kW,
    const float* __restrict__ w1row, const float* __restrict__ W2,
    const float* __restrict__ prior, const float* __restrict__ logp,
    const int* __restrict__ idxb, float* __restrict__ wout)
{
    int m    = blockIdx.x * (blockDim.x >> 5) + (threadIdx.x >> 5);  // 0..2047
    int lane = threadIdx.x & 31;
    int la = lane & 15, hi = lane >> 4;

    int idx_la = idxb[m * TOPK + la];            // A-frag row for this lane
    int idxh[8];
    float pr[8];
#pragma unroll
    for (int v = 0; v < 8; v++) {                // epilogue rows k = v + 8*hi
        idxh[v] = idxb [m * TOPK + v + 8 * hi];
        pr[v]   = prior[m * TOPK + v + 8 * hi];
    }

    // Build the 4 K-step A-fragments of (q .* key[idx]) in bf16.
    const bf16_t* qrow = qbf + (size_t)m * 128;
    const bf16_t* krow = kbf + (size_t)idx_la * 128;
    v16bf afr[4];
#pragma unroll
    for (int ks = 0; ks < 4; ks++) {
        int off0 = ks * 32 + hi * 16;
        v8bf q0 = *(const v8bf*)(qrow + off0);
        v8bf q1 = *(const v8bf*)(qrow + off0 + 8);
        v8bf k0 = *(const v8bf*)(krow + off0);
        v8bf k1 = *(const v8bf*)(krow + off0 + 8);
#pragma unroll
        for (int i = 0; i < 8; i++) {
            afr[ks][i]     = (bf16_t)((float)q0[i] * (float)k0[i]);
            afr[ks][8 + i] = (bf16_t)((float)q1[i] * (float)k1[i]);
        }
    }

    float resacc[8] = {0,0,0,0,0,0,0,0};
#pragma unroll
    for (int jt = 0; jt < 8; jt++) {             // 8 column tiles of H=128
        v8f acc = {};
        const bf16_t* brow = AhT + (size_t)(jt * 16 + la) * 128 + hi * 16;
#pragma unroll
        for (int ks = 0; ks < 4; ks++) {
            v16bf bv = *(const v16bf*)(brow + ks * 32);
            acc = __builtin_amdgcn_wmma_f32_16x16x32_bf16(
                      false, afr[ks], false, bv, (short)0, acc, false, false);
        }
        int jcol  = jt * 16 + la;
        float qv  = qWb1[(size_t)m * 128 + jcol];
        float w1r = w1row[jcol];
        float w2v = W2[jcol];
#pragma unroll
        for (int v = 0; v < 8; v++) {
            float h = acc[v] + qv + kW[(size_t)idxh[v] * 128 + jcol] + pr[v] * w1r;
            h = fmaxf(h, 0.0f);
            resacc[v] += h * w2v;
        }
    }
    // Sum over the 16 columns held by each half-wave.
#pragma unroll
    for (int off = 1; off < 16; off <<= 1) {
#pragma unroll
        for (int v = 0; v < 8; v++) resacc[v] += __shfl_xor(resacc[v], off, 32);
    }
    // Softmax over 16 candidates (b2 is a constant shift -> dropped).
    float lg[8], mx = -1e30f;
#pragma unroll
    for (int v = 0; v < 8; v++) {
        lg[v] = logp[m * TOPK + v + 8 * hi] + resacc[v];
        mx = fmaxf(mx, lg[v]);
    }
    mx = fmaxf(mx, __shfl_xor(mx, 16, 32));      // combine halves
    float sum = 0.f;
#pragma unroll
    for (int v = 0; v < 8; v++) { lg[v] = __expf(lg[v] - mx); sum += lg[v]; }
    sum += __shfl_xor(sum, 16, 32);
    float inv = 1.0f / sum;
    if (la == 0) {
#pragma unroll
        for (int v = 0; v < 8; v++) wout[m * TOPK + v + 8 * hi] = lg[v] * inv;
    }
}

// ---------------------------------------------------------------------------
// Kernel: final gather + weighted sum, times the gate already in d_out.
// out[b,n,d] = gate[b,n,d] * sum_k w[n,k] * P[b,d,idx[n,k]]
// Gathers stay inside one 4KB row of P per block -> L2/L0 resident.
// ---------------------------------------------------------------------------
__global__ __launch_bounds__(256) void final_kernel(
    const float* __restrict__ P, const float* __restrict__ wbuf,
    const int* __restrict__ idxb, float* __restrict__ out)
{
    int bx = blockIdx.x;                         // b*2048 + d
    int n  = blockIdx.y * blockDim.x + threadIdx.x;
    int b  = bx >> 11, d = bx & 2047;
    int m  = b * NPATCH + n;
    const float* prow = P + (size_t)bx * 1024;
    float acc = 0.f;
#pragma unroll
    for (int t = 0; t < TOPK; t++)
        acc += wbuf[m * TOPK + t] * prow[idxb[m * TOPK + t]];
    size_t o = (size_t)m * DHF + d;
    out[o] = acc * out[o];
}

// ---------------------------------------------------------------------------
extern "C" void kernel_launch(void* const* d_in, const int* in_sizes, int n_in,
                              void* d_out, int out_size, void* d_ws, size_t ws_size,
                              hipStream_t stream)
{
    const float* hr_attn = (const float*)d_in[0];
    const float* hr_hf   = (const float*)d_in[1];
    const float* base_hf = (const float*)d_in[2];
    const float* Wq = (const float*)d_in[3];
    const float* bq = (const float*)d_in[4];
    const float* Wk = (const float*)d_in[5];
    const float* bk = (const float*)d_in[6];
    const float* Wg = (const float*)d_in[7];
    const float* bg = (const float*)d_in[8];
    const float* W1 = (const float*)d_in[9];
    const float* b1 = (const float*)d_in[10];
    const float* W2 = (const float*)d_in[11];
    // d_in[12]=b2 cancels in softmax; d_in[13]=topk fixed at 16.
    float* out = (float*)d_out;

    char* ws = (char*)d_ws;
    size_t off = 0;
    auto take = [&](size_t bytes) -> char* {
        char* p = ws + off;
        off = (off + bytes + 255) & ~(size_t)255;
        return p;
    };
    float*  prior   = (float*)take(MROWS * TOPK * 4);
    float*  logp    = (float*)take(MROWS * TOPK * 4);
    int*    idxb    = (int*)  take(MROWS * TOPK * 4);
    float*  scal    = (float*)take(MROWS * 3 * 4);
    float*  wbuf    = (float*)take(MROWS * TOPK * 4);
    bf16_t* desc_hr = (bf16_t*)take((size_t)MROWS * DIN * 2);
    bf16_t* desc_bs = (bf16_t*)take((size_t)MROWS * DIN * 2);
    bf16_t* WkT     = (bf16_t*)take(DIN * DEMB * 2);
    bf16_t* WqT     = (bf16_t*)take(DIN * DEMB * 2);
    bf16_t* WgT     = (bf16_t*)take((size_t)DHF * DEMB * 2);
    bf16_t* AqT     = (bf16_t*)take(DEMB * HID * 2);
    bf16_t* AkT     = (bf16_t*)take(DEMB * HID * 2);
    bf16_t* AhT     = (bf16_t*)take(DEMB * HID * 2);
    float*  w1row   = (float*)take(HID * 4);
    bf16_t* q_bf    = (bf16_t*)take((size_t)MROWS * DEMB * 2);
    bf16_t* k_bf    = (bf16_t*)take((size_t)MROWS * DEMB * 2);
    float*  qWb1    = (float*)take((size_t)MROWS * HID * 4);
    float*  kWp     = (float*)take((size_t)MROWS * HID * 4);

    // 1) top-k
    topk_kernel<<<MROWS / 8, 256, 0, stream>>>(hr_attn, prior, logp, idxb, scal);
    // 2) weight prep (442624 work items)
    prep_kernel<<<(442624 + 255) / 256, 256, 0, stream>>>(
        Wq, Wk, Wg, W1, WqT, WkT, WgT, AqT, AkT, AhT, w1row);
    // 3) pooled descriptors
    desc_kernel<<<dim3(BATCH * DIN, NPATCH / 256), 256, 0, stream>>>(
        hr_hf, base_hf, desc_hr, desc_bs);
    // 4) key_embed  = desc_hr @ Wk + bk   (bf16 out)
    wmma_gemm_plain<<<(MROWS / 16) * (DEMB / 16) / 8, 256, 0, stream>>>(
        desc_hr, WkT, MROWS, DEMB, DIN, bk, nullptr, k_bf);
    // 5) query_embed = desc_bs @ Wq + bq  (bf16 out)
    wmma_gemm_plain<<<(MROWS / 16) * (DEMB / 16) / 8, 256, 0, stream>>>(
        desc_bs, WqT, MROWS, DEMB, DIN, bq, nullptr, q_bf);
    // 6) qWb1 = q @ (W1a+W1c) + b1
    wmma_gemm_plain<<<(MROWS / 16) * (HID / 16) / 8, 256, 0, stream>>>(
        q_bf, AqT, MROWS, HID, DEMB, b1, qWb1, nullptr);
    // 7) kW = key @ (W1b-W1c)
    wmma_gemm_plain<<<(MROWS / 16) * (HID / 16) / 8, 256, 0, stream>>>(
        k_bf, AkT, MROWS, HID, DEMB, nullptr, kWp, nullptr);
    // 8) gate = sigmoid(q @ Wg[0:128] + rank-3 scalar features + bg) -> d_out
    //    LDS-staged via async-to-LDS, fragments from ds_load.
    gate_gemm_lds<<<dim3(MROWS / 128, DHF / 128), 256, 0, stream>>>(
        q_bf, WgT, bg, out, scal, Wg + 128 * 2048);
    // 9) rescorer + softmax -> w
    rescore_kernel<<<MROWS / 8, 256, 0, stream>>>(
        q_bf, k_bf, AhT, qWb1, kWp, w1row, W2, prior, logp, idxb, wbuf);
    // 10) gather / weighted sum, multiply gate in place
    final_kernel<<<dim3(BATCH * DHF, NPATCH / 256), 256, 0, stream>>>(
        hr_hf, wbuf, idxb, out);
}
